// AttentionBlock_77257871721112
// MI455X (gfx1250) — matrix operand order repaired and checked by reading, not compiled, not run on hardware
//
#include <hip/hip_runtime.h>

typedef __attribute__((ext_vector_type(16))) _Float16 v16h;
typedef __attribute__((ext_vector_type(8)))  float    v8f;
typedef __attribute__((ext_vector_type(4)))  int      v4i;
typedef _Float16 half_t;

// Problem constants
#define BATCH 4
#define CCH   512
#define SSEQ  4096
#define HEADS 8
#define HDIM  64
#define BHN   (BATCH*HEADS)

// GEMM tiling
#define BM 128
#define BN 256
#define BK 32
#define ALD 40     // A-tile LDS row stride (halfs): 80B rows (16B aligned), bank-spread
#define BTLD 40    // Bt-tile LDS row stride (halfs)

// Attention tiling
#define AW 4       // waves per block
#define AQ 64      // queries per wave
#define LD 72      // attention LDS row stride (halfs): 144B rows (16B aligned)

// ---------------- async global->LDS (CDNA5), guarded with fallback ----------
#if defined(__has_builtin)
#if __has_builtin(__builtin_amdgcn_global_load_async_to_lds_b128)
#define HAVE_ASYNC_LDS 1
#endif
#endif
#ifndef HAVE_ASYNC_LDS
#define HAVE_ASYNC_LDS 0
#endif

typedef __attribute__((address_space(1))) v4i gas_v4i;
typedef __attribute__((address_space(3))) v4i las_v4i;

__device__ __forceinline__ void copy16_async(half_t* ldst, const half_t* gsrc) {
#if HAVE_ASYNC_LDS
  __builtin_amdgcn_global_load_async_to_lds_b128((gas_v4i*)gsrc, (las_v4i*)ldst, 0, 0);
#else
  *(uint4*)ldst = *(const uint4*)gsrc;
#endif
}

__device__ __forceinline__ void wait_async_lds() {
#if HAVE_ASYNC_LDS
  asm volatile("s_wait_asynccnt 0" ::: "memory");
#endif
}

__device__ __forceinline__ v8f wmma16(v16h a, v16h b, v8f c) {
  return __builtin_amdgcn_wmma_f32_16x16x32_f16(
      false, a, false, b, (short)0, c, false, false);
}

__device__ __forceinline__ unsigned int pack2(float a, float b) {
  union { _Float16 h[2]; unsigned int u; } t;
  t.h[0] = (half_t)a; t.h[1] = (half_t)b;
  return t.u;
}

// A fragment: 16x32 f16 (k = 16*hi + s, contiguous; same order as load_bt_frag); rows contiguous along K
// in LDS -> 2x ds_load_b128 per lane.
__device__ __forceinline__ v16h load_a_frag(const half_t* lds, int row0, int k0, int ld) {
  const int lane = threadIdx.x & 31;
  const int hi   = lane >> 4;
  const int row  = row0 + (lane & 15);
  union { v16h h; unsigned int u[8]; } r;
  const unsigned int* p32 = (const unsigned int*)lds;
  const int base = row * ld + k0;
#pragma unroll
  for (int p = 0; p < 8; ++p) {
    const int kp = 2 * p + 16 * hi;
    r.u[p] = p32[(base + kp) >> 1];
  }
  return r.h;
}

// B fragment from a TRANSPOSED tile Bt[N][K]: 8 contiguous dwords per lane.
__device__ __forceinline__ v16h load_bt_frag(const half_t* lds, int n0, int k0, int ld) {
  const int lane = threadIdx.x & 31;
  const int hi   = lane >> 4;
  const int n    = n0 + (lane & 15);
  union { v16h h; unsigned int u[8]; } r;
  const unsigned int* p32 = (const unsigned int*)lds;
  const int base = (n * ld + k0 + 16 * hi) >> 1;
#pragma unroll
  for (int p = 0; p < 8; ++p) r.u[p] = p32[base + p];
  return r.h;
}

// ---------------------------------------------------------------- utilities
__global__ void cvt_f32_f16_kernel(const float* __restrict__ in,
                                   half_t* __restrict__ out, int n) {
  int i = blockIdx.x * 256 + threadIdx.x;
  if (i < n) out[i] = (half_t)in[i];
}

// -------------------------------------------------------------- group norm
__global__ __launch_bounds__(256) void groupnorm_kernel(
    const float* __restrict__ x, const float* __restrict__ w,
    const float* __restrict__ b, half_t* __restrict__ h16) {
  const int bg = blockIdx.x;
  const int bb = bg >> 5, g = bg & 31;
  const size_t base = ((size_t)bb * CCH + (size_t)g * 16) * SSEQ;
  const int tid = threadIdx.x;
  const float4* xv = (const float4*)(x + base);

  float s = 0.f, s2 = 0.f;
  for (int i = tid; i < 16 * SSEQ / 4; i += 256) {
    float4 v = xv[i];
    s  += v.x + v.y + v.z + v.w;
    s2 += v.x * v.x + v.y * v.y + v.z * v.z + v.w * v.w;
  }
  __shared__ float rs[256], rq[256];
  rs[tid] = s; rq[tid] = s2;
  __syncthreads();
  for (int off = 128; off > 0; off >>= 1) {
    if (tid < off) { rs[tid] += rs[tid + off]; rq[tid] += rq[tid + off]; }
    __syncthreads();
  }
  const float inv_n = 1.f / 65536.f;
  const float mu = rs[0] * inv_n;
  const float var = rq[0] * inv_n - mu * mu;
  const float rstd = rsqrtf(var + 1e-5f);
  for (int i = tid; i < 16 * SSEQ / 4; i += 256) {
    const int c = g * 16 + (i >> 10);
    const float sw = rstd * w[c];
    const float sb = b[c] - mu * sw;
    float4 v = xv[i];
    uint2 o;
    o.x = pack2(v.x * sw + sb, v.y * sw + sb);
    o.y = pack2(v.z * sw + sb, v.w * sw + sb);
    *(uint2*)(h16 + base + (size_t)i * 4) = o;
  }
}

// ------------------------------------------------------------- GEMM core
// C[m0:+128, n0:+256] = A[M,K]*B[K,N]. Double-buffered LDS ping-pong:
// A tile staged with async-to-LDS, B tile transposed via v_perm in VGPRs.
__device__ __forceinline__ void stage_a_tile(const half_t* __restrict__ A, int lda,
                                             int m0, int k0, half_t* As) {
  const int tid = threadIdx.x;
#pragma unroll
  for (int t = 0; t < 2; ++t) {            // 128 rows x 4 chunks / 256 thr
    const int i = tid + t * 256;
    const int row = i >> 2, ch = i & 3;
    copy16_async(As + row * ALD + ch * 8,
                 A + (size_t)(m0 + row) * lda + k0 + ch * 8);
  }
}

__device__ __forceinline__ void load_b_regs(const half_t* __restrict__ B, int ldb,
                                            int n0, int k0,
                                            unsigned int d0[8], unsigned int d1[8]) {
  const int tid = threadIdx.x;
  const unsigned int* Bg = (const unsigned int*)(B + (size_t)k0 * ldb + n0);
#pragma unroll
  for (int t = 0; t < 8; ++t) {            // (BK/2)*(BN/2)/256 = 8
    const int i = tid + t * 256;
    const int kk = i >> 7, np = i & 127;
    d0[t] = Bg[(size_t)(2 * kk + 0) * (ldb >> 1) + np];
    d1[t] = Bg[(size_t)(2 * kk + 1) * (ldb >> 1) + np];
  }
}

__device__ __forceinline__ void store_b_tile(const unsigned int d0[8],
                                             const unsigned int d1[8], half_t* Bt) {
  const int tid = threadIdx.x;
  unsigned int* Bw = (unsigned int*)Bt;
#pragma unroll
  for (int t = 0; t < 8; ++t) {
    const int i = tid + t * 256;
    const int kk = i >> 7, np = i & 127;
    unsigned int o0 = __builtin_amdgcn_perm(d1[t], d0[t], 0x05040100u);
    unsigned int o1 = __builtin_amdgcn_perm(d1[t], d0[t], 0x07060302u);
    Bw[(2 * np + 0) * (BTLD / 2) + kk] = o0;
    Bw[(2 * np + 1) * (BTLD / 2) + kk] = o1;
  }
}

__device__ __forceinline__ void gemm_core(
    const half_t* __restrict__ A, int lda,
    const half_t* __restrict__ B, int ldb,
    half_t* As, half_t* Bt, int m0, int n0, int K, v8f acc[4][4]) {
  const int tid  = threadIdx.x;
  const int wave = tid >> 5;
  const int wm   = wave >> 2;   // 0..1
  const int wn   = wave & 3;    // 0..3

#pragma unroll
  for (int mi = 0; mi < 4; ++mi)
#pragma unroll
    for (int ni = 0; ni < 4; ++ni) acc[mi][ni] = (v8f){};

  unsigned int bd0[8], bd1[8];
  // prologue: stage tile 0 into buffer 0
  stage_a_tile(A, lda, m0, 0, As);
  load_b_regs(B, ldb, n0, 0, bd0, bd1);
  store_b_tile(bd0, bd1, Bt);
  wait_async_lds();
  __syncthreads();

  for (int k0 = 0, it = 0; k0 < K; k0 += BK, ++it) {
    const int cur = it & 1, nxt = cur ^ 1;
    const bool more = (k0 + BK) < K;
    half_t* Ac = As + cur * (BM * ALD);
    half_t* Bc = Bt + cur * (BN * BTLD);

    if (more) {
      stage_a_tile(A, lda, m0, k0 + BK, As + nxt * (BM * ALD));
      load_b_regs(B, ldb, n0, k0 + BK, bd0, bd1);
    }

    v16h af[4], bf[4];
#pragma unroll
    for (int mi = 0; mi < 4; ++mi) af[mi] = load_a_frag(Ac, wm * 64 + mi * 16, 0, ALD);
#pragma unroll
    for (int ni = 0; ni < 4; ++ni) bf[ni] = load_bt_frag(Bc, wn * 64 + ni * 16, 0, BTLD);
#pragma unroll
    for (int mi = 0; mi < 4; ++mi)
#pragma unroll
      for (int ni = 0; ni < 4; ++ni)
        acc[mi][ni] = wmma16(af[mi], bf[ni], acc[mi][ni]);

    if (more) store_b_tile(bd0, bd1, Bt + nxt * (BN * BTLD));
    wait_async_lds();
    __syncthreads();
  }
}

// ------------------------------------------------------------- QKV GEMM
// Layouts produced: q16[bh][s][d] (scaled), k16[bh][s][d], v16[bh][d][s].
__global__ __launch_bounds__(256) void qkv_gemm_kernel(
    const half_t* __restrict__ w16, const half_t* __restrict__ h16,
    const float* __restrict__ qkv_b,
    half_t* __restrict__ q16, half_t* __restrict__ k16, half_t* __restrict__ v16) {
  __shared__ half_t As[2 * BM * ALD];
  __shared__ half_t Bt[2 * BN * BTLD];
  const int b  = blockIdx.z;
  const int m0 = blockIdx.y * BM;
  const int n0 = blockIdx.x * BN;

  v8f acc[4][4];
  gemm_core(w16, CCH, h16 + (size_t)b * CCH * SSEQ, SSEQ, As, Bt, m0, n0, CCH, acc);

  const int lane = threadIdx.x & 31;
  const int wave = threadIdx.x >> 5;
  const int wm = wave >> 2, wn = wave & 3;
  const int hi = lane >> 4, lc = lane & 15;
  const float scale = 0.125f;  // 1/sqrt(64)

#pragma unroll
  for (int mi = 0; mi < 4; ++mi)
#pragma unroll
    for (int ni = 0; ni < 4; ++ni) {
      const int obase = m0 + wm * 64 + mi * 16 + 8 * hi;   // +v, v=0..7
      const int s     = n0 + wn * 64 + ni * 16 + lc;
      float vals[8];
#pragma unroll
      for (int v = 0; v < 8; ++v) vals[v] = acc[mi][ni][v] + qkv_b[obase + v];
      const int which = obase >> 9;          // uniform over v
      const int c0 = obase & 511;
      const size_t bh = (size_t)(b * HEADS + (c0 >> 6));
      const int dd0 = c0 & 63;
      if (which == 0) {
        unsigned int* dst = (unsigned int*)(q16 + ((bh * SSEQ + s) * HDIM + dd0));
#pragma unroll
        for (int t = 0; t < 4; ++t)
          dst[t] = pack2(vals[2 * t] * scale, vals[2 * t + 1] * scale);
      } else if (which == 1) {
        unsigned int* dst = (unsigned int*)(k16 + ((bh * SSEQ + s) * HDIM + dd0));
#pragma unroll
        for (int t = 0; t < 4; ++t)
          dst[t] = pack2(vals[2 * t], vals[2 * t + 1]);
      } else {
        // v16[bh][d][s]: pack along s via lane pairs, even lane stores dword
#pragma unroll
        for (int v = 0; v < 8; ++v) {
          unsigned int me = pack2(vals[v], 0.f) & 0xffffu;
          int other = __shfl_xor((int)me, 1, 32);
          if ((lc & 1) == 0) {
            unsigned int dw = me | ((unsigned int)other << 16);
            *(unsigned int*)(v16 + ((bh * HDIM + dd0 + v) * SSEQ + s)) = dw;
          }
        }
      }
    }
}

// ------------------------------------------------------- flash attention
__global__ __launch_bounds__(128) void attn_kernel(
    const half_t* __restrict__ q16, const half_t* __restrict__ k16,
    const half_t* __restrict__ v16, half_t* __restrict__ attn16) {
  __shared__ half_t kt[64 * LD];        // [j][d]
  __shared__ half_t vt[64 * LD];        // [d][j]
  __shared__ half_t pt[AW][64 * LD];    // per-wave [row][k]

  const int bh   = blockIdx.y;
  const int b    = bh >> 3, hd = bh & 7;
  const int tid  = threadIdx.x;
  const int wave = tid >> 5;
  const int lane = tid & 31;
  const int hi = lane >> 4, lc = lane & 15;
  const int qi0 = blockIdx.x * (AW * AQ) + wave * AQ;

  const half_t* qp = q16 + (size_t)bh * SSEQ * HDIM;   // [s][d]
  const half_t* kp = k16 + (size_t)bh * SSEQ * HDIM;   // [s][d]
  const half_t* vp = v16 + (size_t)bh * HDIM * SSEQ;   // [d][s]

  // stage this wave's Q tile (64x64) via async copies
  for (int i = lane; i < 64 * 8; i += 32) {
    int row = i >> 3, ch = i & 7;                      // 8x 16B chunks per row
    copy16_async(pt[wave] + row * LD + ch * 8,
                 qp + (size_t)(qi0 + row) * HDIM + ch * 8);
  }
  wait_async_lds();
  __syncthreads();
  v16h qf[4][2];
#pragma unroll
  for (int mi = 0; mi < 4; ++mi)
#pragma unroll
    for (int ks = 0; ks < 2; ++ks)
      qf[mi][ks] = load_a_frag(pt[wave], mi * 16, ks * 32, LD);

  v8f O[4][4];
#pragma unroll
  for (int mi = 0; mi < 4; ++mi)
#pragma unroll
    for (int nd = 0; nd < 4; ++nd) O[mi][nd] = (v8f){};
  float m_r[4][8], l_r[4][8];
#pragma unroll
  for (int mi = 0; mi < 4; ++mi)
#pragma unroll
    for (int v = 0; v < 8; ++v) { m_r[mi][v] = -1e30f; l_r[mi][v] = 0.f; }

  for (int j0 = 0; j0 < SSEQ; j0 += 64) {
    __syncthreads();
    // K tile: rows j (contiguous d); V tile: rows d (contiguous j)
    for (int i = tid; i < 64 * 8; i += 128) {
      int row = i >> 3, ch = i & 7;
      copy16_async(kt + row * LD + ch * 8,
                   kp + (size_t)(j0 + row) * HDIM + ch * 8);
    }
    for (int i = tid; i < 64 * 8; i += 128) {
      int row = i >> 3, ch = i & 7;
      copy16_async(vt + row * LD + ch * 8,
                   vp + (size_t)row * SSEQ + j0 + ch * 8);
    }
    wait_async_lds();
    __syncthreads();

    // scores = Q . K^T  (Bt = kt[j][d]: N=j, K=d)
    v8f sc[4][4];
#pragma unroll
    for (int mi = 0; mi < 4; ++mi)
#pragma unroll
      for (int ni = 0; ni < 4; ++ni) sc[mi][ni] = (v8f){};
#pragma unroll
    for (int ks = 0; ks < 2; ++ks) {
      v16h bf[4];
#pragma unroll
      for (int ni = 0; ni < 4; ++ni) bf[ni] = load_bt_frag(kt, ni * 16, ks * 32, LD);
#pragma unroll
      for (int mi = 0; mi < 4; ++mi)
#pragma unroll
        for (int ni = 0; ni < 4; ++ni)
          sc[mi][ni] = wmma16(qf[mi][ks], bf[ni], sc[mi][ni]);
    }

    // online softmax over this 64-key block
#pragma unroll
    for (int mi = 0; mi < 4; ++mi)
#pragma unroll
      for (int v = 0; v < 8; ++v) {
        float mx = fmaxf(fmaxf(sc[mi][0][v], sc[mi][1][v]),
                         fmaxf(sc[mi][2][v], sc[mi][3][v]));
        for (int off = 8; off > 0; off >>= 1)
          mx = fmaxf(mx, __shfl_xor(mx, off, 32));
        const float mnew = fmaxf(m_r[mi][v], mx);
        const float alpha = __expf(m_r[mi][v] - mnew);
        m_r[mi][v] = mnew;
        float ls = 0.f;
#pragma unroll
        for (int ni = 0; ni < 4; ++ni) {
          float p = __expf(sc[mi][ni][v] - mnew);
          sc[mi][ni][v] = p;
          ls += p;
        }
        for (int off = 8; off > 0; off >>= 1)
          ls += __shfl_xor(ls, off, 32);
        l_r[mi][v] = l_r[mi][v] * alpha + ls;
#pragma unroll
        for (int nd = 0; nd < 4; ++nd) O[mi][nd][v] *= alpha;
      }

    // P: D-layout -> A-layout in pt[wave], dword-packed via lane pairs
    half_t* pw = pt[wave];
#pragma unroll
    for (int mi = 0; mi < 4; ++mi)
#pragma unroll
      for (int ni = 0; ni < 4; ++ni)
#pragma unroll
        for (int v = 0; v < 8; ++v) {
          unsigned int me = pack2(sc[mi][ni][v], 0.f) & 0xffffu;
          int other = __shfl_xor((int)me, 1, 32);
          if ((lc & 1) == 0) {
            unsigned int dw = me | ((unsigned int)other << 16);
            *(unsigned int*)(pw + (mi * 16 + v + 8 * hi) * LD + ni * 16 + lc) = dw;
          }
        }
    asm volatile("s_wait_dscnt 0" ::: "memory");

    // O += P @ V   (Bt = vt[d][j]: N=d, K=j)
#pragma unroll
    for (int ks = 0; ks < 2; ++ks) {
      v16h pa[4], vb[4];
#pragma unroll
      for (int mi = 0; mi < 4; ++mi) pa[mi] = load_a_frag(pw, mi * 16, ks * 32, LD);
#pragma unroll
      for (int nd = 0; nd < 4; ++nd) vb[nd] = load_bt_frag(vt, nd * 16, ks * 32, LD);
#pragma unroll
      for (int mi = 0; mi < 4; ++mi)
#pragma unroll
        for (int nd = 0; nd < 4; ++nd)
          O[mi][nd] = wmma16(pa[mi], vb[nd], O[mi][nd]);
    }
  }

  // epilogue: attn16[b][hd*64+d][s] = O[srow][dcol]/l  (dword packed over rows)
  half_t* ap = attn16 + ((size_t)b * CCH + (size_t)hd * HDIM) * SSEQ;
#pragma unroll
  for (int mi = 0; mi < 4; ++mi)
#pragma unroll
    for (int nd = 0; nd < 4; ++nd) {
      const int srow0 = qi0 + mi * 16 + 8 * hi;
      const int dcol  = nd * 16 + lc;
#pragma unroll
      for (int t = 0; t < 4; ++t) {
        float a = O[mi][nd][2 * t + 0] / l_r[mi][2 * t + 0];
        float c = O[mi][nd][2 * t + 1] / l_r[mi][2 * t + 1];
        *(unsigned int*)(ap + ((size_t)dcol * SSEQ + srow0 + 2 * t)) = pack2(a, c);
      }
    }
}

// ------------------------------------------------------------- proj GEMM
__global__ __launch_bounds__(256) void proj_gemm_kernel(
    const half_t* __restrict__ w16, const half_t* __restrict__ attn16,
    const float* __restrict__ proj_b, const float* __restrict__ x,
    float* __restrict__ out) {
  __shared__ half_t As[2 * BM * ALD];
  __shared__ half_t Bt[2 * BN * BTLD];
  const int b  = blockIdx.z;
  const int m0 = blockIdx.y * BM;
  const int n0 = blockIdx.x * BN;

  v8f acc[4][4];
  gemm_core(w16, CCH, attn16 + (size_t)b * CCH * SSEQ, SSEQ, As, Bt, m0, n0, CCH, acc);

  const int lane = threadIdx.x & 31;
  const int wave = threadIdx.x >> 5;
  const int wm = wave >> 2, wn = wave & 3;
  const int hi = lane >> 4, lc = lane & 15;

#pragma unroll
  for (int mi = 0; mi < 4; ++mi)
#pragma unroll
    for (int ni = 0; ni < 4; ++ni)
#pragma unroll
      for (int v = 0; v < 8; ++v) {
        const int o = m0 + wm * 64 + mi * 16 + v + 8 * hi;
        const int s = n0 + wn * 64 + ni * 16 + lc;
        const size_t idx = ((size_t)b * CCH + o) * SSEQ + s;
        out[idx] = acc[mi][ni][v] + proj_b[o] + x[idx];
      }
}

// ---------------------------------------------------------------- launch
extern "C" void kernel_launch(void* const* d_in, const int* in_sizes, int n_in,
                              void* d_out, int out_size, void* d_ws, size_t ws_size,
                              hipStream_t stream) {
  const float* x      = (const float*)d_in[0];
  const float* norm_w = (const float*)d_in[1];
  const float* norm_b = (const float*)d_in[2];
  const float* qkv_w  = (const float*)d_in[3];
  const float* qkv_b  = (const float*)d_in[4];
  const float* proj_w = (const float*)d_in[5];
  const float* proj_b = (const float*)d_in[6];
  float* out = (float*)d_out;

  char* ws = (char*)d_ws;
  const size_t TEN = (size_t)BATCH * CCH * SSEQ * sizeof(half_t);   // 16 MiB
  half_t* h16    = (half_t*)ws;                 ws += TEN;
  half_t* q16    = (half_t*)ws;                 ws += TEN;
  half_t* k16    = (half_t*)ws;                 ws += TEN;
  half_t* v16    = (half_t*)ws;                 ws += TEN;
  half_t* attn16 = (half_t*)ws;                 ws += TEN;
  half_t* qw16   = (half_t*)ws;                 ws += (size_t)3 * CCH * CCH * sizeof(half_t);
  half_t* pw16   = (half_t*)ws;                 ws += (size_t)CCH * CCH * sizeof(half_t);

  const int nqw = 3 * CCH * CCH;
  const int npw = CCH * CCH;
  cvt_f32_f16_kernel<<<(nqw + 255) / 256, 256, 0, stream>>>(qkv_w, qw16, nqw);
  cvt_f32_f16_kernel<<<(npw + 255) / 256, 256, 0, stream>>>(proj_w, pw16, npw);

  groupnorm_kernel<<<BATCH * 32, 256, 0, stream>>>(x, norm_w, norm_b, h16);

  qkv_gemm_kernel<<<dim3(SSEQ / BN, (3 * CCH) / BM, BATCH), 256, 0, stream>>>(
      qw16, h16, qkv_b, q16, k16, v16);

  attn_kernel<<<dim3(SSEQ / (AW * AQ), BHN), AW * 32, 0, stream>>>(
      q16, k16, v16, attn16);

  proj_gemm_kernel<<<dim3(SSEQ / BN, CCH / BM, BATCH), 256, 0, stream>>>(
      pw16, attn16, proj_b, x, out);
}
